// Net_43696997269963
// MI455X (gfx1250) — 8-bit matrix operand order repaired and checked by reading, compiled, not run on hardware
//
#include <hip/hip_runtime.h>
#include <stdint.h>


typedef __attribute__((ext_vector_type(16))) int   v16i;
typedef __attribute__((ext_vector_type(8)))  float v8f;

#define FP8_P1 0x38u
#define FP8_M1 0xB8u

static __device__ __forceinline__ unsigned char fp8_sign(float v) {
  return v > 0.f ? (unsigned char)FP8_P1 : (v < 0.f ? (unsigned char)FP8_M1 : (unsigned char)0);
}

__global__ void k_binarize_input(const float* __restrict__ x,
                                 float* __restrict__ im,
                                 unsigned char* __restrict__ act0,
                                 int total  ) {
  int idx = blockIdx.x * blockDim.x + threadIdx.x;
  if (idx >= total) return;
  int n = idx >> 10, p = idx & 1023;
  const float* xp = x + (size_t)n * 3072 + p;
  unsigned pack = 0;
#pragma unroll
  for (int c = 0; c < 3; ++c) {
    float v = xp[c << 10];
    im[(size_t)n * 3072 + (c << 10) + p] = v > 0.f ? 1.f : (v < 0.f ? -1.f : 0.f);
    pack |= ((unsigned)fp8_sign(v)) << (8 * c);
  }
  *(unsigned*)(act0 + ((size_t)idx << 2)) = pack;
}

__global__ void k_quant_weights(const float* __restrict__ w,
                                unsigned char* __restrict__ wq,
                                int Co, int Ci_w, int Ci_eff, int Kpad) {
  int idx = blockIdx.x * blockDim.x + threadIdx.x;
  if (idx >= Co * Kpad) return;
  int co = idx / Kpad, k = idx - co * Kpad;
  unsigned char code = 0;
  int Kreal = 9 * Ci_eff;
  if (k < Kreal) {
    int tap = k / Ci_eff, c = k - tap * Ci_eff;
    if (c < Ci_w) code = fp8_sign(w[((size_t)co * Ci_w + c) * 9 + tap]);
  }
  wq[idx] = code;
}

__global__ void k_im2col(const unsigned char* __restrict__ act,
                         unsigned* __restrict__ A,
                         int H, int W, int Ci, int Kpad, int Kreal, int M) {
  int kq = Kpad >> 2;
  long long total = (long long)M * kq;
  long long idx = (long long)blockIdx.x * blockDim.x + threadIdx.x;
  if (idx >= total) return;
  int pos = (int)(idx / kq);
  int kd  = (int)(idx - (long long)pos * kq) << 2;
  unsigned val = 0;
  if (kd < Kreal) {
    int tap = kd / Ci, c = kd - tap * Ci;
    int ky = tap / 3, kx = tap - ky * 3;
    int hw = H * W;
    int n = pos / hw, rem = pos - n * hw;
    int y = rem / W, xo = rem - y * W;
    int iy = y + ky - 1, ix = xo + kx - 1;
    if (iy >= 0 && iy < H && ix >= 0 && ix < W)
      val = *(const unsigned*)(act + (size_t)((n * H + iy) * W + ix) * Ci + c);
  }
  A[idx] = val;
}

template <int NT>
__global__ __launch_bounds__(256) void k_gemm_fp8(
    const unsigned char* __restrict__ A,
    const unsigned char* __restrict__ Wq,
    float* __restrict__ C,
    int Co, int Kpad) {
  int wave = threadIdx.x >> 5;
  int lane = threadIdx.x & 31;
  int wm = wave & 3, wn = wave >> 2;
  int m0 = blockIdx.x * 128 + wm * 32;
  int n0 = blockIdx.y * (2 * NT * 16) + wn * (NT * 16);
  int lrow = lane & 15, lhi = lane >> 4;

  const unsigned char* Ar0 = A + (size_t)(m0 + lrow)      * Kpad;
  const unsigned char* Ar1 = A + (size_t)(m0 + 16 + lrow) * Kpad;
  const unsigned char* Br[NT];
#pragma unroll
  for (int t = 0; t < NT; ++t)
    Br[t] = Wq + (size_t)(n0 + t * 16 + lrow) * Kpad;

  v8f acc[2][NT];
#pragma unroll
  for (int i = 0; i < 2; ++i)
#pragma unroll
    for (int t = 0; t < NT; ++t) { v8f z = {}; acc[i][t] = z; }

  for (int k = 0; k < Kpad; k += 128) {
    __builtin_prefetch(Ar0 + k + 128, 0, 0);
    __builtin_prefetch(Ar1 + k + 128, 0, 0);
    v16i a0, a1, b[NT];
#pragma unroll
    for (int v = 0; v < 16; ++v) {
      int offA = ((v >> 3) << 6) + (((v >> 1) & 3) << 4) + (lhi << 3) + ((v & 1) << 2);
      a0[v] = *(const int*)(Ar0 + k + offA);
      a1[v] = *(const int*)(Ar1 + k + offA);
    }
#pragma unroll
    for (int t = 0; t < NT; ++t)
#pragma unroll
      for (int v = 0; v < 16; ++v) {
        int offB = ((v >> 3) << 6) + (((v >> 1) & 3) << 4) + (lhi << 3) + ((v & 1) << 2);
        b[t][v] = *(const int*)(Br[t] + k + offB);
      }
#pragma unroll
    for (int t = 0; t < NT; ++t) {
      acc[0][t] = __builtin_amdgcn_wmma_f32_16x16x128_fp8_fp8(a0, b[t], (short)0, acc[0][t], false, false);
      acc[1][t] = __builtin_amdgcn_wmma_f32_16x16x128_fp8_fp8(a1, b[t], (short)0, acc[1][t], false, false);
    }
  }

#pragma unroll
  for (int mi = 0; mi < 2; ++mi)
#pragma unroll
    for (int t = 0; t < NT; ++t)
#pragma unroll
      for (int r = 0; r < 8; ++r) {
        int m = m0 + mi * 16 + (lhi << 3) + r;
        int n = n0 + t * 16 + lrow;
        C[(size_t)m * Co + n] = acc[mi][t][r];
      }
}

__global__ void k_zero(float* __restrict__ p, int n) {
  int i = blockIdx.x * blockDim.x + threadIdx.x;
  if (i < n) p[i] = 0.f;
}

__global__ void k_bn_reduce(const float* __restrict__ Cm, float* __restrict__ stats,
                            int M, int Co, int parts) {
  int idx = blockIdx.x * blockDim.x + threadIdx.x;
  if (idx >= Co * parts) return;
  int c = idx % Co, part = idx / Co;
  float s = 0.f, q = 0.f;
  for (int m = part; m < M; m += parts) {
    float v = Cm[(size_t)m * Co + c];
    s += v; q += v * v;
  }
  atomicAdd(&stats[c], s);
  atomicAdd(&stats[Co + c], q);
}

__global__ void k_bn_finalize(float* __restrict__ stats, const float* __restrict__ g,
                              const float* __restrict__ b, int Co, int M) {
  int c = blockIdx.x * blockDim.x + threadIdx.x;
  if (c >= Co) return;
  float inv  = 1.f / (float)M;
  float mean = stats[c] * inv;
  float var  = stats[Co + c] * inv - mean * mean;
  var = var > 0.f ? var : 0.f;
  float sc = g[c] * rsqrtf(var + 1e-5f);
  stats[2 * Co + c] = sc;
  stats[3 * Co + c] = b[c] - mean * sc;
}

__global__ void k_bn_pool_bin(const float* __restrict__ Cm, const float* __restrict__ stats,
                              unsigned char* __restrict__ actOut,
                              int Hin, int Win, int Co, int pool, int total) {
  int idx = blockIdx.x * blockDim.x + threadIdx.x;
  if (idx >= total) return;
  int c = idx % Co, t = idx / Co;
  int Wo = pool ? (Win >> 1) : Win;
  int Ho = pool ? (Hin >> 1) : Hin;
  int xo = t % Wo; t /= Wo;
  int yo = t % Ho; int n = t / Ho;
  float sc = stats[2 * Co + c], sh = stats[3 * Co + c];
  float best;
  if (pool) {
    int y0 = yo << 1, x0 = xo << 1;
    size_t base = ((size_t)(n * Hin + y0) * Win + x0) * Co + c;
    size_t rs = (size_t)Win * Co;
    float v00 = Cm[base] * sc + sh;
    float v01 = Cm[base + Co] * sc + sh;
    float v10 = Cm[base + rs] * sc + sh;
    float v11 = Cm[base + rs + Co] * sc + sh;
    best = fmaxf(fmaxf(v00, v01), fmaxf(v10, v11));
  } else {
    best = Cm[((size_t)(n * Hin + yo) * Win + xo) * Co + c] * sc + sh;
  }
  best = fmaxf(best, 0.f);
  actOut[idx] = best > 0.f ? (unsigned char)FP8_P1 : (unsigned char)0;
}

__global__ void k_fc1(const unsigned char* __restrict__ act, const float* __restrict__ fw,
                      const float* __restrict__ fb, float* __restrict__ out) {
  int idx = blockIdx.x * blockDim.x + threadIdx.x;
  if (idx >= 256 * 512) return;
  int bq = idx / 512, j = idx - bq * 512;
  const float* wr = fw + (size_t)j * 512;
  const unsigned char* ar = act + (size_t)bq * 512;
  float s = fb[j];
  for (int c = 0; c < 512; ++c) {
    unsigned char code = ar[c];
    float a = (code == FP8_P1) ? 1.f : (code == FP8_M1 ? -1.f : 0.f);
    s = fmaf(a, wr[c], s);
  }
  out[idx] = s;
}

__global__ void k_cls(const float* __restrict__ h, const float* __restrict__ cw,
                      const float* __restrict__ cb, float* __restrict__ out) {
  int idx = blockIdx.x * blockDim.x + threadIdx.x;
  if (idx >= 256 * 10) return;
  int bq = idx / 10, k = idx - bq * 10;
  const float* wr = cw + (size_t)k * 512;
  const float* hr = h + (size_t)bq * 512;
  float s = cb[k];
  for (int j = 0; j < 512; ++j) s = fmaf(hr[j], wr[j], s);
  out[idx] = s;
}

#define WS_A      ((size_t)0)
#define WS_C      ((size_t)41943040)
#define WS_ACT0   ((size_t)109051904)
#define WS_ACT1   ((size_t)113246208)
#define WS_WQ     ((size_t)117440512)
#define WS_STATS  ((size_t)119799808)
#define WS_FC1    ((size_t)119808000)

extern "C" void kernel_launch(void* const* d_in, const int* in_sizes, int n_in,
                              void* d_out, int out_size, void* d_ws, size_t ws_size,
                              hipStream_t stream) {
  (void)in_sizes; (void)n_in; (void)out_size; (void)ws_size;

  const float* x = (const float*)d_in[0];
  float* out_logits = (float*)d_out;
  float* out_im     = (float*)d_out + 256 * 10;

  char* ws = (char*)d_ws;
  unsigned char* bufA  = (unsigned char*)(ws + WS_A);
  float*         bufC  = (float*)(ws + WS_C);
  unsigned char* act[2] = { (unsigned char*)(ws + WS_ACT0), (unsigned char*)(ws + WS_ACT1) };
  unsigned char* wq    = (unsigned char*)(ws + WS_WQ);
  float*         stats = (float*)(ws + WS_STATS);
  float*         fc1   = (float*)(ws + WS_FC1);

  {
    int total = 256 * 1024;
    k_binarize_input<<<(total + 255) / 256, 256, 0, stream>>>(x, out_im, act[0], total);
  }

  struct LayerCfg { int ciw, cie, co, hin, pool; };
  const LayerCfg Ls[8] = {
    {  3,   4,  64, 32, 1},
    { 64,  64, 128, 16, 1},
    {128, 128, 256,  8, 0},
    {256, 256, 256,  8, 1},
    {256, 256, 512,  4, 0},
    {512, 512, 512,  4, 1},
    {512, 512, 512,  2, 0},
    {512, 512, 512,  2, 1},
  };

  int cur = 0;
  for (int i = 0; i < 8; ++i) {
    const LayerCfg L = Ls[i];
    int H = L.hin, W = L.hin;
    int M = 256 * H * W;
    int Kreal = 9 * L.cie;
    int Kpad  = (Kreal + 127) & ~127;
    const float* w = (const float*)d_in[1 + 3 * i];
    const float* g = (const float*)d_in[2 + 3 * i];
    const float* b = (const float*)d_in[3 + 3 * i];

    int wtot = L.co * Kpad;
    k_quant_weights<<<(wtot + 255) / 256, 256, 0, stream>>>(w, wq, L.co, L.ciw, L.cie, Kpad);

    long long dtot = (long long)M * (Kpad >> 2);
    k_im2col<<<(unsigned)((dtot + 255) / 256), 256, 0, stream>>>(
        act[cur], (unsigned*)bufA, H, W, L.cie, Kpad, Kreal, M);

    if (L.co % 128 == 0) {
      dim3 gg(M / 128, L.co / 128);
      k_gemm_fp8<4><<<gg, 256, 0, stream>>>(bufA, wq, bufC, L.co, Kpad);
    } else {
      dim3 gg(M / 128, L.co / 64);
      k_gemm_fp8<2><<<gg, 256, 0, stream>>>(bufA, wq, bufC, L.co, Kpad);
    }

    k_zero<<<(2 * L.co + 255) / 256, 256, 0, stream>>>(stats, 2 * L.co);
    const int parts = 256;
    k_bn_reduce<<<(L.co * parts + 255) / 256, 256, 0, stream>>>(bufC, stats, M, L.co, parts);
    k_bn_finalize<<<(L.co + 255) / 256, 256, 0, stream>>>(stats, g, b, L.co, M);

    int Ho = L.pool ? H / 2 : H, Wo = L.pool ? W / 2 : W;
    int ototal = 256 * Ho * Wo * L.co;
    k_bn_pool_bin<<<(ototal + 255) / 256, 256, 0, stream>>>(
        bufC, stats, act[cur ^ 1], H, W, L.co, L.pool, ototal);
    cur ^= 1;
  }

  k_fc1<<<(256 * 512 + 255) / 256, 256, 0, stream>>>(
      act[cur], (const float*)d_in[25], (const float*)d_in[26], fc1);
  k_cls<<<(256 * 10 + 255) / 256, 256, 0, stream>>>(
      fc1, (const float*)d_in[27], (const float*)d_in[28], out_logits);
}
